// CorrLSTM_resize_norm_29609504539006
// MI455X (gfx1250) — matrix operand order repaired and checked by reading, not compiled, not run on hardware
//
#include <hip/hip_runtime.h>

typedef __attribute__((ext_vector_type(16))) __bf16 v16bf;
typedef __attribute__((ext_vector_type(8)))  __bf16 v8bf;
typedef __attribute__((ext_vector_type(8)))  float  v8f;
typedef __attribute__((ext_vector_type(4)))  unsigned v4u;
typedef __attribute__((ext_vector_type(8)))  unsigned v8u;

#define PSZ 11
#define PP  121
#define NB  8

// ---- workspace layout (bytes) ----
#define WS_CORR (0ull)            // 15.9 MB fp32 corr (level0 max)
#define WS_INP  (16ull << 20)     // 16 MB bf16 (B,256,H,H)
#define WS_WB   (32ull << 20)     // 2.36 MB bf16 [512][2304]
#define WS_CONV (36ull << 20)     // 64 MB fp32 [512][Ntot]
#define WS_MAX  (100ull << 20)    // 4 B ordered-uint max

// ---- helpers ----
__device__ __forceinline__ unsigned f2key(float f) {
  int b = __float_as_int(f);
  return (b >= 0) ? (unsigned(b) | 0x80000000u) : ~unsigned(b);
}
__device__ __forceinline__ float key2f(unsigned k) {
  int b = (k & 0x80000000u) ? int(k & 0x7FFFFFFFu) : ~int(k);
  return __int_as_float(b);
}
__device__ __forceinline__ __bf16 f2bf(float f) {
  unsigned u = __float_as_uint(f);
  unsigned r = (u + 0x7FFFu + ((u >> 16) & 1u)) >> 16;
  unsigned short s = (unsigned short)r;
  __bf16 out;
  __builtin_memcpy(&out, &s, 2);
  return out;
}

__global__ void init_max_kernel(unsigned* mk) { *mk = 0u; }

// ================= correlation + leaky + global max =================
__global__ __launch_bounds__(64) void corr_kernel(
    const float* __restrict__ f1, const float* __restrict__ f2,
    float* __restrict__ corr, unsigned* __restrict__ maxkey,
    int C, int H) {
  const int W = H;
  const int WP = W + 10;
  const int by = blockIdx.x;
  const int b = by / H;
  const int y = by % H;
  const int tid = threadIdx.x;
  const int x = tid;

  __shared__ float sf2[4 * 11 * 74];
  __shared__ float smax[64];

  float acc[PP];
#pragma unroll
  for (int t = 0; t < PP; ++t) acc[t] = 0.f;

  const int wcount = 11 * WP;

  for (int c0 = 0; c0 < C; c0 += 4) {
    for (int i = tid; i < 4 * wcount; i += 64) {
      int cc  = i / wcount;
      int rem = i - cc * wcount;
      int r   = rem / WP;
      int col = rem - r * WP;
      int yy = y + r - 5;
      int xx = col - 5;
      float v = 0.f;
      if (yy >= 0 && yy < H && xx >= 0 && xx < W)
        v = f2[((size_t)(b * C + c0 + cc) * H + yy) * W + xx];
      sf2[i] = v;
    }
    __syncthreads();
    if (x < W) {
#pragma unroll
      for (int cc = 0; cc < 4; ++cc) {
        float f1v = f1[((size_t)(b * C + c0 + cc) * H + y) * W + x];
        const float* wrow = &sf2[cc * wcount];
#pragma unroll
        for (int i = 0; i < PSZ; ++i) {
#pragma unroll
          for (int j = 0; j < PSZ; ++j)
            acc[i * PSZ + j] = fmaf(f1v, wrow[i * WP + x + j], acc[i * PSZ + j]);
        }
      }
    }
    __syncthreads();
  }

  float lmax = -3.4e38f;
  if (x < W) {
#pragma unroll
    for (int t = 0; t < PP; ++t) {
      float v = acc[t];
      v = v > 0.f ? v : 0.01f * v;             // LeakyReLU(0.01)
      corr[((size_t)(b * PP + t) * H + y) * W + x] = v;
      lmax = fmaxf(lmax, v);
    }
  }
  smax[tid] = lmax;
  __syncthreads();
#pragma unroll
  for (int s = 32; s > 0; s >>= 1) {
    if (tid < s) smax[tid] = fmaxf(smax[tid], smax[tid + s]);
    __syncthreads();
  }
  if (tid == 0) atomicMax(maxkey, f2key(smax[0]));
}

// ================= pack normalized corr + h into bf16 (B,256,H,H) =================
__global__ void pack_inp_kernel(const float* __restrict__ corr,
                                const float* __restrict__ hpre,
                                const unsigned* __restrict__ maxkey,
                                __bf16* __restrict__ inp, int H) {
  const int spatial = H * H;
  size_t idx = (size_t)blockIdx.x * 256 + threadIdx.x;
  size_t total = (size_t)NB * 256 * spatial;
  if (idx >= total) return;
  int s = (int)(idx % spatial);
  int c = (int)((idx / spatial) % 256);
  int b = (int)(idx / ((size_t)spatial * 256));
  float v = 0.f;
  if (c < PP) {
    float inv = 1.f / key2f(*maxkey);
    v = corr[((size_t)(b * PP + c)) * spatial + s] * inv;
  } else if (c < 2 * PP) {
    v = hpre[((size_t)(b * PP + (c - PP))) * spatial + s];
  }
  inp[idx] = f2bf(v);
}

// ================= pack weights to bf16 [512][9*256] =================
__global__ void pack_w_kernel(const float* __restrict__ W, __bf16* __restrict__ Wb) {
  int idx = blockIdx.x * 256 + threadIdx.x;
  if (idx >= 512 * 9 * 256) return;
  int ci  = idx & 255;
  int tap = (idx >> 8) % 9;
  int oc  = idx / (9 * 256);
  float v = 0.f;
  if (oc < 4 * PP && ci < 2 * PP)
    v = W[((size_t)(oc * (2 * PP) + ci)) * 9 + tap];
  Wb[idx] = f2bf(v);
}

// ================= implicit-GEMM 3x3 conv via WMMA bf16 + TDM A-tiles =================
// D[m,n] = sum_k A[m,k]*B[k,n]; M=512 (oc pad), N=B*H*H, K=9*256=2304.
// Block: 256 thr / 8 waves, 64x64 output tile, K staged 64 per barrier round.
// A (weight) tile staged by the Tensor Data Mover; B tile is an im2col gather.
__global__ __launch_bounds__(256) void conv_gemm_kernel(
    const __bf16* __restrict__ Wb, const __bf16* __restrict__ inp,
    float* __restrict__ convout, int H) {
  const int spatial = H * H;
  const int Ntot = NB * spatial;
  const int mBase = blockIdx.x * 64;
  const int nBase = blockIdx.y * 64;
  const int tid  = threadIdx.x;
  const int lane = tid & 31;
  const int wave = tid >> 5;
  const int wm = wave & 3;
  const int wn = wave >> 2;
  const int lm = lane & 15;
  const int kh = lane >> 4;

  // single LDS block so sA sits at LDS offset 0 (used as D# lds_addr)
  __shared__ __bf16 smem[2 * 64 * 64];
  __bf16* sA = smem;            // [m][k]  64x64
  __bf16* sB = smem + 64 * 64;  // [n][k]  64x64 (transposed for frag reads)

  v8f acc0 = {};
  v8f acc1 = {};

  // B staging: fixed k (channel), 16 consecutive pixels per thread
  const int bko = tid >> 2;           // 0..63 (k within 64-chunk)
  const int bng = (tid & 3) * 16;     // 0,16,32,48 (pixel group)

  // Hoist pixel decomposition out of the K loop (loop-invariant per block)
  int basei[16], yc[16], xc[16];
#pragma unroll
  for (int j = 0; j < 16; ++j) {
    int n = nBase + bng + j;
    if (n < Ntot) {
      int bb = n / spatial;
      int s  = n - bb * spatial;
      int y  = s / H;
      basei[j] = bb * 256 * spatial + s;  // index at channel 0, tap (0,0)
      yc[j] = y;
      xc[j] = s - y * H;
    } else {
      basei[j] = 0;
      yc[j] = -1000;
      xc[j] = -1000;
    }
  }

  // ---- Tensor DMA descriptor group1 (uniform across K loop) ----
  // data_size=2B, tensor 512x2304 bf16, tile 64x64, row stride 2304.
  const v8u g1 = { 0x00010000u,                 // workgroup_mask=0, data_size=1(2B)
                   0x09000000u,                 // tensor_dim0 = 2304 (bits 79:48)
                   0x02000000u,                 // tensor_dim1 = 512  (bits 111:80)
                   (unsigned)(64 << 16),        // tile_dim0 = 64     (bits 127:112)
                   64u,                         // tile_dim1 = 64     (bits 143:128)
                   2304u,                       // tensor_dim0_stride (bits 207:160)
                   0u, 0u };
  const unsigned long long wbBase = (unsigned long long)Wb;

  for (int kc = 0; kc < 36; ++kc) {           // 36 rounds of K=64
    const int tap   = kc >> 2;                // 4 chunks of 64 per tap (256 ch)
    const int cbase = (kc & 3) * 64;
    const int dy = tap / 3 - 1;
    const int dx = tap % 3 - 1;

    // ---- stage A tile via Tensor Data Mover (one wave issues) ----
    if (wave == 0) {
      unsigned long long ga =
          wbBase + ((unsigned long long)mBase * 2304ull + (unsigned long long)kc * 64ull) * 2ull;
      v4u g0 = { 1u,                               // count=1, user desc
                 0u,                               // lds_addr = sA (LDS base)
                 (unsigned)ga,                     // global_addr[31:0]
                 ((unsigned)(ga >> 32) & 0x01FFFFFFu) | 0x80000000u }; // addr[56:32] | type=2
      asm volatile("tensor_load_to_lds %0, %1" :: "s"(g0), "s"(g1) : "memory");
      __builtin_amdgcn_s_wait_tensorcnt(0);
    }

    // ---- stage B tile (implicit im2col gather, borders zero) ----
    {
      const int ch = cbase + bko;
      const size_t choff = (size_t)ch * spatial + (dy * H + dx);
#pragma unroll
      for (int j = 0; j < 16; ++j) {
        __bf16 v = f2bf(0.f);
        int yy = yc[j] + dy;
        int xx = xc[j] + dx;
        if (((unsigned)yy < (unsigned)H) & ((unsigned)xx < (unsigned)H))
          v = inp[choff + basei[j]];
        sB[(bng + j) * 64 + bko] = v;
      }
    }
    __syncthreads();

    const __bf16* pa  = &sA[(wm * 16 + lm) * 64];
    const __bf16* pb0 = &sB[(wn * 32 + lm) * 64];
    const __bf16* pb1 = &sB[(wn * 32 + 16 + lm) * 64];

#pragma unroll
    for (int s = 0; s < 2; ++s) {             // two K=32 steps per stage
      const int ko = s * 32;
      // A frag: K = ko + kh*16 .. +15 contiguous
      v8bf alo = *(const v8bf*)(pa + ko + kh * 16);
      v8bf ahi = *(const v8bf*)(pa + ko + kh * 16 + 8);
      v16bf a = __builtin_shufflevector(alo, ahi, 0, 1, 2, 3, 4, 5, 6, 7,
                                        8, 9, 10, 11, 12, 13, 14, 15);
      // B frags: K = ko + kh*16 .. +15 contiguous
      v8bf b0lo = *(const v8bf*)(pb0 + ko + kh * 16);
      v8bf b0hi = *(const v8bf*)(pb0 + ko + kh * 16 + 8);
      v16bf bf0 = __builtin_shufflevector(b0lo, b0hi, 0, 1, 2, 3, 4, 5, 6, 7,
                                          8, 9, 10, 11, 12, 13, 14, 15);
      v8bf b1lo = *(const v8bf*)(pb1 + ko + kh * 16);
      v8bf b1hi = *(const v8bf*)(pb1 + ko + kh * 16 + 8);
      v16bf bf1 = __builtin_shufflevector(b1lo, b1hi, 0, 1, 2, 3, 4, 5, 6, 7,
                                          8, 9, 10, 11, 12, 13, 14, 15);

      acc0 = __builtin_amdgcn_wmma_f32_16x16x32_bf16(false, a, false, bf0,
                                                     (short)0, acc0, false, false);
      acc1 = __builtin_amdgcn_wmma_f32_16x16x32_bf16(false, a, false, bf1,
                                                     (short)0, acc1, false, false);
    }
    __syncthreads();
  }

  // D layout: VGPR r -> M = r + 8*kh, N = lm
#pragma unroll
  for (int r = 0; r < 8; ++r) {
    int m  = mBase + wm * 16 + kh * 8 + r;
    int n0 = nBase + wn * 32 + lm;
    int n1 = n0 + 16;
    if (n0 < Ntot) convout[(size_t)m * Ntot + n0] = acc0[r];
    if (n1 < Ntot) convout[(size_t)m * Ntot + n1] = acc1[r];
  }
}

// ================= fused LSTM pointwise =================
__global__ void lstm_kernel(const float* __restrict__ conv,
                            const float* __restrict__ bias,
                            const float* __restrict__ cpre,
                            float* __restrict__ hout, float* __restrict__ cout,
                            int H) {
  const int spatial = H * H;
  const int Ntot = NB * spatial;
  size_t idx = (size_t)blockIdx.x * 256 + threadIdx.x;
  if (idx >= (size_t)PP * Ntot) return;
  int n = (int)(idx % Ntot);
  int c = (int)(idx / Ntot);
  int b = n / spatial;
  int s = n - b * spatial;

  float ci = conv[(size_t)c * Ntot + n] + bias[c];
  float cf = conv[(size_t)(PP + c) * Ntot + n] + bias[PP + c];
  float co = conv[(size_t)(2 * PP + c) * Ntot + n] + bias[2 * PP + c];
  float cg = conv[(size_t)(3 * PP + c) * Ntot + n] + bias[3 * PP + c];
  float ig = 1.f / (1.f + __expf(-ci));
  float fg = 1.f / (1.f + __expf(-cf));
  float og = 1.f / (1.f + __expf(-co));
  float gg = tanhf(cg);
  float cp = cpre[((size_t)(b * PP + c)) * spatial + s];
  float cn = fg * cp + ig * gg;
  float hn = og * tanhf(cn);
  size_t o = ((size_t)(b * PP + c)) * spatial + s;
  hout[o] = hn;
  cout[o] = cn;
}

// ================= host launcher =================
extern "C" void kernel_launch(void* const* d_in, const int* in_sizes, int n_in,
                              void* d_out, int out_size, void* d_ws, size_t ws_size,
                              hipStream_t stream) {
  (void)in_sizes; (void)n_in; (void)out_size; (void)ws_size;
  static const int Hs[6] = {64, 32, 16, 8, 4, 2};
  static const int Cs[6] = {512, 1024, 512, 256, 256, 256};

  char* ws = (char*)d_ws;
  float*    corr  = (float*)(ws + WS_CORR);
  __bf16*   inp   = (__bf16*)(ws + WS_INP);
  __bf16*   Wb    = (__bf16*)(ws + WS_WB);
  float*    convO = (float*)(ws + WS_CONV);
  unsigned* mk    = (unsigned*)(ws + WS_MAX);
  float*    out   = (float*)d_out;

  size_t ooff = 0;
  for (int L = 0; L < 6; ++L) {
    const int H = Hs[L], C = Cs[L];
    const int spatial = H * H;
    const int Ntot = NB * spatial;
    const size_t Nel = (size_t)NB * PP * spatial;

    const float* x  = (const float*)d_in[6 * L + 0];
    const float* xp = (const float*)d_in[6 * L + 1];
    const float* h  = (const float*)d_in[6 * L + 2];
    const float* cp = (const float*)d_in[6 * L + 3];
    const float* Wt = (const float*)d_in[6 * L + 4];
    const float* bs = (const float*)d_in[6 * L + 5];

    init_max_kernel<<<1, 1, 0, stream>>>(mk);
    corr_kernel<<<NB * H, 64, 0, stream>>>(x, xp, corr, mk, C, H);

    size_t npack = (size_t)NB * 256 * spatial;
    pack_inp_kernel<<<(unsigned)((npack + 255) / 256), 256, 0, stream>>>(corr, h, mk, inp, H);
    pack_w_kernel<<<(512 * 9 * 256) / 256, 256, 0, stream>>>(Wt, Wb);

    dim3 gg(8, (unsigned)((Ntot + 63) / 64));
    conv_gemm_kernel<<<gg, 256, 0, stream>>>(Wb, inp, convO, H);

    size_t nl = (size_t)PP * Ntot;
    lstm_kernel<<<(unsigned)((nl + 255) / 256), 256, 0, stream>>>(
        convO, bs, cp, out + ooff, out + ooff + Nel, H);

    ooff += 2 * Nel;
  }
}
